// MultiHeadAttention_13855564497508
// MI455X (gfx1250) — matrix operand order repaired and checked by reading, not compiled, not run on hardware
//
#include <hip/hip_runtime.h>

// ---------------------------------------------------------------------------
// MI455X (gfx1250) fused multi-head attention.
// bf16 WMMA (v_wmma_f32_16x16x32_bf16) + f32 accumulate.
// Double-buffered global_load_async_to_lds_b128 (ASYNCcnt) staging overlaps
// HBM->LDS DMA with matrix math. All fragment loads are ds_load_b128 pairs.
// ---------------------------------------------------------------------------

typedef __bf16 bf16;
typedef __bf16  v16bf  __attribute__((ext_vector_type(16)));
typedef __bf16  bf16x8 __attribute__((ext_vector_type(8)));
typedef float   v8f    __attribute__((ext_vector_type(8)));

#define HEADS  12
#define DMODEL 768
#define HDIM   64
#define SEQ    1024
#define BATCH  8
#define MROWS  (BATCH * SEQ)   // 8192
#define LDSP   72              // padded LDS row stride (bf16), 144B: 16B-aligned

__device__ __forceinline__ v16bf frag_cat(bf16x8 lo, bf16x8 hi) {
  return __builtin_shufflevector(lo, hi, 0, 1, 2, 3, 4, 5, 6, 7, 8, 9, 10, 11,
                                 12, 13, 14, 15);
}

// A-frag (16-bit A 16x32): lane half h -> elems 0..7 = K h*8+0..7,
// elems 8..15 = K 16+h*8+0..7  => two contiguous 16B ds_load_b128.
__device__ __forceinline__ v16bf load_a_frag(const bf16* p, int half) {
  bf16x8 lo = *(const bf16x8*)(p + half * 8);
  bf16x8 hi = *(const bf16x8*)(p + 16 + half * 8);
  return frag_cat(lo, hi);
}

// B-frag (16-bit B 32x16): lane = half*16+n, elems 0..7 -> K = half*8+0..7,
// elems 8..15 -> K = 16+half*8+0..7 (matches A-frag).
// p points at the K-contiguous row for column n => two 16B ds_load_b128.
__device__ __forceinline__ v16bf load_b_frag(const bf16* p, int half) {
  bf16x8 lo = *(const bf16x8*)(p + half * 8);
  bf16x8 hi = *(const bf16x8*)(p + 16 + half * 8);
  return frag_cat(lo, hi);
}

__device__ __forceinline__ v8f wmma_bf16(v16bf a, v16bf b, v8f c) {
  return __builtin_amdgcn_wmma_f32_16x16x32_bf16(false, a, false, b, (short)0, c,
                                                 false, false);
}

__device__ __forceinline__ v8f zero8() {
  v8f z;
#pragma unroll
  for (int i = 0; i < 8; ++i) z[i] = 0.0f;
  return z;
}

// gfx1250 async DMA: global -> LDS, 16 bytes per lane, tracked by ASYNCcnt.
__device__ __forceinline__ void async_copy_b128(void* lds_dst, const void* gsrc) {
  unsigned lds_off = (unsigned)(unsigned long long)lds_dst;  // low 32b = LDS addr
  asm volatile("global_load_async_to_lds_b128 %0, %1, off"
               :
               : "v"(lds_off), "v"(gsrc)
               : "memory");
}
// Async loads retire in order: wait "<= N remaining" retires older tile's ops.
__device__ __forceinline__ void async_wait_0()  { asm volatile("s_wait_asynccnt 0x0" ::: "memory"); }
__device__ __forceinline__ void async_wait_8()  { asm volatile("s_wait_asynccnt 0x8" ::: "memory"); }
__device__ __forceinline__ void async_wait_12() { asm volatile("s_wait_asynccnt 0xc" ::: "memory"); }

// ---------------------------------------------------------------------------
// fp32 -> bf16 (grid-stride)
// ---------------------------------------------------------------------------
__global__ __launch_bounds__(256) void f32_to_bf16_kernel(
    const float* __restrict__ in, bf16* __restrict__ out, int n) {
  int i = blockIdx.x * 256 + threadIdx.x;
  int stride = gridDim.x * 256;
  for (; i < n; i += stride) out[i] = (bf16)in[i];
}

// ---------------------------------------------------------------------------
// fp32 [R][C] -> bf16 transposed [C][R]  (weights: K-contiguous B operand)
// ---------------------------------------------------------------------------
__global__ __launch_bounds__(256) void transpose_f32_to_bf16_kernel(
    const float* __restrict__ in, bf16* __restrict__ out, int R, int C) {
  __shared__ float t[32][33];
  int bx = blockIdx.x * 32, by = blockIdx.y * 32;
  int tx = threadIdx.x & 31, ty = threadIdx.x >> 5;  // 32 x 8
  for (int yy = ty; yy < 32; yy += 8)
    t[yy][tx] = in[(size_t)(by + yy) * C + bx + tx];
  __syncthreads();
  for (int yy = ty; yy < 32; yy += 8)
    out[(size_t)(bx + yy) * R + by + tx] = (bf16)t[tx][yy];
}

// ---------------------------------------------------------------------------
// GEMM: C = A[MxK] * W + bias; A row-major bf16, Wt = W^T [N][K] bf16.
//   mode 0: bf16 head-major      [B,H,S,64], scaled by outScale   (Q,K)
//   mode 2: bf16 head-transposed [B,H,64,S]                       (V)
//   mode 1: f32 row-major [M,N]                                   (out)
// Block 128 thr (4 waves); wave tile 32x64; block tile 128x64.
// Double-buffered async staging: 12 async ops / thread / k-tile.
// ---------------------------------------------------------------------------
__global__ __launch_bounds__(128) void gemm_bf16_kernel(
    const bf16* __restrict__ A, const bf16* __restrict__ Wt,
    const float* __restrict__ bias, int M, int N, int K,
    float outScale, int mode,
    bf16* __restrict__ outB, float* __restrict__ outF32) {
  __shared__ bf16 lA[2][128 * LDSP];
  __shared__ bf16 lB[2][64 * LDSP];

  const int tid  = threadIdx.x;
  const int lane = tid & 31;
  const int wave = tid >> 5;
  const int ln16 = lane & 15;
  const int half = lane >> 4;

  const int rowBase = blockIdx.y * 128;
  const int colBase = blockIdx.x * 64;

  // issue one 64-wide k-tile into buffer `buf` (uniform, fully unrolled)
  auto issue_tile = [&](int kt, int buf) {
#pragma unroll
    for (int k = 0; k < 8; ++k) {           // A tile: 128x64 = 1024 b128 chunks
      int i = tid + k * 128;
      int r = i >> 3, c = (i & 7) << 3;
      async_copy_b128(&lA[buf][r * LDSP + c],
                      &A[(size_t)(rowBase + r) * K + kt + c]);
    }
#pragma unroll
    for (int k = 0; k < 4; ++k) {           // Wt tile: 64x64 = 512 b128 chunks
      int i = tid + k * 128;
      int r = i >> 3, c = (i & 7) << 3;
      async_copy_b128(&lB[buf][r * LDSP + c],
                      &Wt[(size_t)(colBase + r) * K + kt + c]);
    }
  };

  v8f acc[2][4];
#pragma unroll
  for (int am = 0; am < 2; ++am)
#pragma unroll
    for (int s = 0; s < 4; ++s) acc[am][s] = zero8();

  issue_tile(0, 0);
  int buf = 0;
  for (int kt = 0; kt < K; kt += 64, buf ^= 1) {
    if (kt + 64 < K) {          // prefetch next tile, retire current tile's DMA
      issue_tile(kt + 64, buf ^ 1);
      async_wait_12();
    } else {
      async_wait_0();
    }
    __syncthreads();

#pragma unroll
    for (int ks = 0; ks < 2; ++ks) {
      v16bf aF[2];
#pragma unroll
      for (int am = 0; am < 2; ++am)
        aF[am] = load_a_frag(
            &lA[buf][(wave * 32 + am * 16 + ln16) * LDSP + ks * 32], half);
#pragma unroll
      for (int sub = 0; sub < 4; ++sub) {
        v16bf b =
            load_b_frag(&lB[buf][(sub * 16 + ln16) * LDSP + ks * 32], half);
#pragma unroll
        for (int am = 0; am < 2; ++am)
          acc[am][sub] = wmma_bf16(aF[am], b, acc[am][sub]);
      }
    }
    __syncthreads();
  }

  // epilogue: C/D layout row m = half*8 + r, col n = lane&15
#pragma unroll
  for (int am = 0; am < 2; ++am) {
    const int row0 = rowBase + wave * 32 + am * 16 + half * 8;
#pragma unroll
    for (int sub = 0; sub < 4; ++sub) {
      int col = colBase + sub * 16 + ln16;
      float bcol = bias[col];
      if (mode == 1) {
#pragma unroll
        for (int r = 0; r < 8; ++r)
          outF32[(size_t)(row0 + r) * N + col] = acc[am][sub][r] + bcol;
      } else {
        int h = col >> 6, dd = col & 63;
#pragma unroll
        for (int r = 0; r < 8; ++r) {
          int row = row0 + r;
          int bb = row / SEQ, ss = row % SEQ;
          float val = (acc[am][sub][r] + bcol) * outScale;
          size_t idx = (mode == 0)
                           ? ((size_t)(bb * HEADS + h) * SEQ + ss) * HDIM + dd
                           : ((size_t)(bb * HEADS + h) * HDIM + dd) * SEQ + ss;
          outB[idx] = (bf16)val;
        }
      }
    }
  }
}

// ---------------------------------------------------------------------------
// Flash attention per (b,h). Q,K head-major [B*H,S,64]; V head-transposed
// [B*H,64,S]; Q pre-scaled by 1/sqrt(d). Ctx written bf16 row-major [B,S,D].
// Block 128 thr (4 waves); 64 queries per block, 16 per wave.
// Double-buffered async K/V staging: 8 async ops / thread / key-tile.
// ---------------------------------------------------------------------------
__global__ __launch_bounds__(128) void flash_attn_kernel(
    const bf16* __restrict__ Q, const bf16* __restrict__ Kh,
    const bf16* __restrict__ Vt, bf16* __restrict__ Ctx) {
  __shared__ bf16 lQ[64 * LDSP];      // Q tile; reused as P scratch
  __shared__ bf16 lK[2][64 * LDSP];   // [key][dd]
  __shared__ bf16 lV[2][64 * LDSP];   // [dd][key]

  const int tid  = threadIdx.x;
  const int lane = tid & 31;
  const int wave = tid >> 5;
  const int ln16 = lane & 15;
  const int half = lane >> 4;

  const int qt = blockIdx.x;          // query tile (0..15)
  const int bh = blockIdx.y;          // 0..B*H-1
  const int b  = bh / HEADS, h = bh % HEADS;
  const size_t headBase = (size_t)bh * SEQ * HDIM;

  auto issue_kv = [&](int kt, int buf) {
#pragma unroll
    for (int k = 0; k < 4; ++k) {     // K tile [key][dd]: 512 b128 chunks
      int i = tid + k * 128;
      int r = i >> 3, c = (i & 7) << 3;
      async_copy_b128(&lK[buf][r * LDSP + c],
                      &Kh[headBase + (size_t)(kt * 64 + r) * HDIM + c]);
    }
#pragma unroll
    for (int k = 0; k < 4; ++k) {     // V tile [dd][key]: 512 b128 chunks
      int i = tid + k * 128;
      int r = i >> 3, c = (i & 7) << 3;
      async_copy_b128(&lV[buf][r * LDSP + c],
                      &Vt[headBase + (size_t)r * SEQ + kt * 64 + c]);
    }
  };

  // async-stage Q tile (4 ops/thread), then key-tile 0 (8 ops/thread)
#pragma unroll
  for (int k = 0; k < 4; ++k) {
    int i = tid + k * 128;
    int r = i >> 3, c = (i & 7) << 3;
    async_copy_b128(&lQ[r * LDSP + c],
                    &Q[headBase + (size_t)(qt * 64 + r) * HDIM + c]);
  }
  issue_kv(0, 0);
  async_wait_8();        // in-order retirement => the 4 Q ops are complete
  __syncthreads();

  v16bf aQ[2];
#pragma unroll
  for (int ks = 0; ks < 2; ++ks)
    aQ[ks] = load_a_frag(&lQ[(wave * 16 + ln16) * LDSP + ks * 32], half);
  // No barrier needed: each wave's later P writes hit only its own Q rows.

  v8f acc[4];
#pragma unroll
  for (int s = 0; s < 4; ++s) acc[s] = zero8();
  float mrow[8], lrow[8];
#pragma unroll
  for (int r = 0; r < 8; ++r) { mrow[r] = -3.0e38f; lrow[r] = 0.0f; }

  const int NT = SEQ / 64;
  int buf = 0;
  for (int kt = 0; kt < NT; ++kt, buf ^= 1) {
    if (kt + 1 < NT) {   // prefetch next K/V tile, retire current tile's DMA
      issue_kv(kt + 1, buf ^ 1);
      async_wait_8();
    } else {
      async_wait_0();
    }
    __syncthreads();

    // scores S = Q K^T (Q pre-scaled); B elem (K=dd, n=key) = lK[key][dd]
    v8f sc[4];
#pragma unroll
    for (int s = 0; s < 4; ++s) sc[s] = zero8();
#pragma unroll
    for (int ks = 0; ks < 2; ++ks)
#pragma unroll
      for (int sub = 0; sub < 4; ++sub) {
        v16bf bk =
            load_b_frag(&lK[buf][(sub * 16 + ln16) * LDSP + ks * 32], half);
        sc[sub] = wmma_bf16(aQ[ks], bk, sc[sub]);
      }

    // online softmax; row stats per C/D slot, reduced across 16-lane halves
#pragma unroll
    for (int r = 0; r < 8; ++r) {
      float v0 = fmaxf(fmaxf(sc[0][r], sc[1][r]), fmaxf(sc[2][r], sc[3][r]));
      v0 = fmaxf(v0, __shfl_xor(v0, 1, 32));
      v0 = fmaxf(v0, __shfl_xor(v0, 2, 32));
      v0 = fmaxf(v0, __shfl_xor(v0, 4, 32));
      v0 = fmaxf(v0, __shfl_xor(v0, 8, 32));
      float mnew  = fmaxf(mrow[r], v0);
      float alpha = __expf(mrow[r] - mnew);
      mrow[r] = mnew;
      lrow[r] *= alpha;
#pragma unroll
      for (int sub = 0; sub < 4; ++sub) acc[sub][r] *= alpha;
      float ps = 0.0f;
#pragma unroll
      for (int sub = 0; sub < 4; ++sub) {
        float p = __expf(sc[sub][r] - mnew);
        sc[sub][r] = p;
        ps += p;
      }
      ps += __shfl_xor(ps, 1, 32);
      ps += __shfl_xor(ps, 2, 32);
      ps += __shfl_xor(ps, 4, 32);
      ps += __shfl_xor(ps, 8, 32);
      lrow[r] += ps;
    }

    // P -> LDS (per-wave region of lQ). Same-wave store->load: the LDS
    // pipeline is in-order per wave, no workgroup barrier required.
#pragma unroll
    for (int sub = 0; sub < 4; ++sub)
#pragma unroll
      for (int r = 0; r < 8; ++r)
        lQ[(wave * 16 + half * 8 + r) * LDSP + sub * 16 + ln16] =
            (bf16)sc[sub][r];

    // ctx += P @ V ; B elem (K=key, n=dd) = lV[dd][key] (K-contiguous)
#pragma unroll
    for (int ks = 0; ks < 2; ++ks) {
      v16bf aP = load_a_frag(&lQ[(wave * 16 + ln16) * LDSP + ks * 32], half);
#pragma unroll
      for (int sub = 0; sub < 4; ++sub) {
        v16bf bv =
            load_b_frag(&lV[buf][(sub * 16 + ln16) * LDSP + ks * 32], half);
        acc[sub] = wmma_bf16(aP, bv, acc[sub]);
      }
    }
    __syncthreads();   // all waves done with lK/lV[buf] before DMA reuses it
  }

  // normalize and store ctx -> bf16 row-major [B,S,D], col = h*64 + dd
#pragma unroll
  for (int r = 0; r < 8; ++r) {
    float inv = 1.0f / lrow[r];
    int ss = qt * 64 + wave * 16 + half * 8 + r;
    size_t rowOff = ((size_t)b * SEQ + ss) * DMODEL + h * HDIM;
#pragma unroll
    for (int sub = 0; sub < 4; ++sub)
      Ctx[rowOff + sub * 16 + ln16] = (bf16)(acc[sub][r] * inv);
  }
}

// ---------------------------------------------------------------------------
extern "C" void kernel_launch(void* const* d_in, const int* in_sizes, int n_in,
                              void* d_out, int out_size, void* d_ws,
                              size_t ws_size, hipStream_t stream) {
  (void)in_sizes; (void)n_in; (void)out_size; (void)ws_size;
  const float* x  = (const float*)d_in[0];
  const float* Wq = (const float*)d_in[1];
  const float* bq = (const float*)d_in[2];
  const float* Wk = (const float*)d_in[3];
  const float* bk = (const float*)d_in[4];
  const float* Wv = (const float*)d_in[5];
  const float* bv = (const float*)d_in[6];
  const float* Wo = (const float*)d_in[7];
  const float* bo = (const float*)d_in[8];

  char* ws = (char*)d_ws;
  size_t off = 0;
  auto alloc = [&](size_t bytes) -> char* {
    char* p = ws + off;
    off += (bytes + 255) & ~(size_t)255;
    return p;
  };

  const size_t xElems = (size_t)MROWS * DMODEL;
  const size_t wElems = (size_t)DMODEL * DMODEL;

  bf16* xbf = (bf16*)alloc(xElems * 2);  // reused for ctx after projections
  bf16* wqt = (bf16*)alloc(wElems * 2);  // W^T bf16 [N][K]
  bf16* wkt = (bf16*)alloc(wElems * 2);
  bf16* wvt = (bf16*)alloc(wElems * 2);
  bf16* wot = (bf16*)alloc(wElems * 2);
  bf16* qbf = (bf16*)alloc(xElems * 2);  // [B,H,S,64]
  bf16* kbf = (bf16*)alloc(xElems * 2);  // [B,H,S,64]
  bf16* vbf = (bf16*)alloc(xElems * 2);  // [B,H,64,S]
  bf16* ctxbf = xbf;

  f32_to_bf16_kernel<<<dim3(2048), 256, 0, stream>>>(x, xbf, (int)xElems);
  dim3 tgrid(DMODEL / 32, DMODEL / 32);  // (24,24)
  transpose_f32_to_bf16_kernel<<<tgrid, 256, 0, stream>>>(Wq, wqt, DMODEL, DMODEL);
  transpose_f32_to_bf16_kernel<<<tgrid, 256, 0, stream>>>(Wk, wkt, DMODEL, DMODEL);
  transpose_f32_to_bf16_kernel<<<tgrid, 256, 0, stream>>>(Wv, wvt, DMODEL, DMODEL);
  transpose_f32_to_bf16_kernel<<<tgrid, 256, 0, stream>>>(Wo, wot, DMODEL, DMODEL);

  dim3 ggrid(DMODEL / 64, MROWS / 128);  // (12, 64)
  gemm_bf16_kernel<<<ggrid, 128, 0, stream>>>(xbf, wqt, bq, MROWS, DMODEL,
                                              DMODEL, 0.125f, 0, qbf, nullptr);
  gemm_bf16_kernel<<<ggrid, 128, 0, stream>>>(xbf, wkt, bk, MROWS, DMODEL,
                                              DMODEL, 1.0f, 0, kbf, nullptr);
  gemm_bf16_kernel<<<ggrid, 128, 0, stream>>>(xbf, wvt, bv, MROWS, DMODEL,
                                              DMODEL, 1.0f, 2, vbf, nullptr);

  flash_attn_kernel<<<dim3(SEQ / 64, BATCH * HEADS), 128, 0, stream>>>(
      qbf, kbf, vbf, ctxbf);

  gemm_bf16_kernel<<<ggrid, 128, 0, stream>>>(ctxbf, wot, bo, MROWS, DMODEL,
                                              DMODEL, 1.0f, 1, nullptr,
                                              (float*)d_out);
}
